// MLAttention_37933151158584
// MI455X (gfx1250) — compile-verified, matrix operand order checked by reading, not run on hardware
//
#include <hip/hip_runtime.h>
#include <hip/hip_bf16.h>
#include <math.h>

// ---------------------------------------------------------------------------
// MLA forward for MI455X (gfx1250, wave32, WMMA).
// Strategy: bf16 WMMA (v_wmma_f32_16x16x32_bf16) with f32 accumulation for all
// GEMMs; full logits materialized (reference returns attn_weights), masked
// softmax applied in place on the d_out attn_weights region.
// ---------------------------------------------------------------------------

#define BB    2
#define SS    2048
#define HID   2048
#define NH    16
#define Q_LORA  1536
#define KV_LORA 512
#define D_ROPE  64
#define D_NOPE  128
#define D_QK    192
#define D_V     128

typedef float v8f  __attribute__((ext_vector_type(8)));
typedef __bf16 v16bf __attribute__((ext_vector_type(16)));
typedef __bf16 bf16x8 __attribute__((ext_vector_type(8)));

__device__ __forceinline__ __bf16 f2bf(float f) {
    union { float f; unsigned u; } a; a.f = f;
    unsigned u = a.u;
    unsigned r = u + 0x7FFFu + ((u >> 16) & 1u);   // round-to-nearest-even
    unsigned short h = (unsigned short)(r >> 16);
    return __builtin_bit_cast(__bf16, h);
}

// ---------------------------------------------------------------------------
// Generic batched GEMM: C[M,N] = alpha * A[M,K] @ B, f32 in/out, bf16 WMMA.
// b_is_NK == 0 : B stored row-major [K,N] (element (k,n) at B[k*ldb+n])
// b_is_NK == 1 : B stored row-major [N,K] (element (k,n) at B[n*ldb+k]) == B^T
// Block tile 64x64, 256 threads (8 wave32 waves), K-step 32.
// Wave w -> 16x32 strip: tiles (w>>1, (w&1)*32) and (w>>1, (w&1)*32+16).
// ---------------------------------------------------------------------------
__global__ __launch_bounds__(256) void gemm_bf16_wmma(
    const float* __restrict__ A, const float* __restrict__ Bm,
    float* __restrict__ C,
    int M, int N, int K, int lda, int ldb, int ldc,
    long long sA_batch, long long sB_batch, long long sC_batch,
    int b_is_NK, float alpha)
{
    __shared__ __align__(16) __bf16 sA[64 * 32];
    __shared__ __align__(16) __bf16 sB[64 * 32];   // stored [n][k]

    const int tid  = threadIdx.x;
    const int lane = tid & 31;
    const int wid  = tid >> 5;
    const int wm   = wid >> 1;            // 0..3  (M tile row)
    const int wn   = (wid & 1) * 32;      // 0/32  (N strip base)

    const int bm = blockIdx.y * 64;
    const int bn = blockIdx.x * 64;

    const float* Ab = A  + (long long)blockIdx.z * sA_batch;
    const float* Bb = Bm + (long long)blockIdx.z * sB_batch;
    float*       Cb = C  + (long long)blockIdx.z * sC_batch;

    v8f acc0 = {};
    v8f acc1 = {};

    const int frow  = lane & 15;          // fragment row/col within tile
    const int khalf = (lane >> 4) * 8;    // 0 or 8 (ISA 16-bit A/B layout)

    for (int k0 = 0; k0 < K; k0 += 32) {
        // ---- stage A tile 64x32 (f32 -> bf16) ----
        {
            const int r  = tid >> 2;          // 0..63
            const int kc = (tid & 3) * 8;     // 0,8,16,24
            const float* src = Ab + (long long)(bm + r) * lda + (k0 + kc);
            __bf16* dst = &sA[r * 32 + kc];
            #pragma unroll
            for (int j = 0; j < 8; ++j) dst[j] = f2bf(src[j]);
            if (k0 + 32 < K) __builtin_prefetch(src + 32, 0, 0);   // global_prefetch_b8
        }
        // ---- stage B tile as sB[n][k] (f32 -> bf16, transposing if [K,N]) ----
        if (b_is_NK) {
            const int n  = tid >> 2;
            const int kc = (tid & 3) * 8;
            const float* src = Bb + (long long)(bn + n) * ldb + (k0 + kc);
            __bf16* dst = &sB[n * 32 + kc];
            #pragma unroll
            for (int j = 0; j < 8; ++j) dst[j] = f2bf(src[j]);
            if (k0 + 32 < K) __builtin_prefetch(src + 32, 0, 0);
        } else {
            const int kl = tid >> 3;          // 0..31
            const int nc = (tid & 7) * 8;     // 0..56
            const float* src = Bb + (long long)(k0 + kl) * ldb + (bn + nc);
            #pragma unroll
            for (int j = 0; j < 8; ++j) sB[(nc + j) * 32 + kl] = f2bf(src[j]);
            if (k0 + 32 < K) __builtin_prefetch(src + (long long)32 * ldb, 0, 0);
        }
        __syncthreads();

        // ---- build fragments per ISA 16-bit 16x32 layout (two b128 LDS loads) ----
        union { v16bf v; bf16x8 h[2]; } fa, fb0, fb1;
        fa.h[0]  = *(const bf16x8*)&sA[(wm * 16 + frow) * 32 + khalf];
        fa.h[1]  = *(const bf16x8*)&sA[(wm * 16 + frow) * 32 + khalf + 16];
        fb0.h[0] = *(const bf16x8*)&sB[(wn +      frow) * 32 + khalf];
        fb0.h[1] = *(const bf16x8*)&sB[(wn +      frow) * 32 + khalf + 16];
        fb1.h[0] = *(const bf16x8*)&sB[(wn + 16 + frow) * 32 + khalf];
        fb1.h[1] = *(const bf16x8*)&sB[(wn + 16 + frow) * 32 + khalf + 16];

        acc0 = __builtin_amdgcn_wmma_f32_16x16x32_bf16(
                   false, fa.v, false, fb0.v, (short)0, acc0, false, false);
        acc1 = __builtin_amdgcn_wmma_f32_16x16x32_bf16(
                   false, fa.v, false, fb1.v, (short)0, acc1, false, false);

        __syncthreads();
    }

    // ---- store: C/D layout: VGPR r -> M=r (lanes 0-15) / M=8+r (lanes 16-31)
    const int m_base = bm + wm * 16 + (lane >> 4) * 8;
    const int n0     = bn + wn + (lane & 15);
    #pragma unroll
    for (int r = 0; r < 8; ++r) {
        Cb[(long long)(m_base + r) * ldc + n0]      = alpha * acc0[r];
        Cb[(long long)(m_base + r) * ldc + n0 + 16] = alpha * acc1[r];
    }
}

// ---------------------------------------------------------------------------
// RMS norm, one block per row (supports in-place, strided rows for ckv slice)
// ---------------------------------------------------------------------------
__global__ __launch_bounds__(256) void rmsnorm_kernel(
    const float* x, const float* w, float* y,
    int width, int xstride, int ystride)
{
    const long long row = blockIdx.x;
    const float* xr = x + row * xstride;
    float*       yr = y + row * ystride;
    __shared__ float red[256];
    float s = 0.f;
    for (int i = threadIdx.x; i < width; i += 256) { float v = xr[i]; s += v * v; }
    red[threadIdx.x] = s; __syncthreads();
    for (int off = 128; off > 0; off >>= 1) {
        if ((int)threadIdx.x < off) red[threadIdx.x] += red[threadIdx.x + off];
        __syncthreads();
    }
    const float inv = rsqrtf(red[0] / (float)width + 1e-6f);
    for (int i = threadIdx.x; i < width; i += 256) yr[i] = w[i] * (xr[i] * inv);
}

// ---------------------------------------------------------------------------
// Build q_full [B*H,S,192], k_full [B*H,S,192], v [B*H,S,128] with RoPE.
// grid = (S, B)
// ---------------------------------------------------------------------------
__global__ __launch_bounds__(256) void build_qkv_kernel(
    const float* __restrict__ qg,    // [B*S, H*192]
    const float* __restrict__ kvg,   // [B*S, H*256]
    const float* __restrict__ ckv,   // [B*S, 576]  (last 64 = k_pe, pre-rope)
    const float* __restrict__ cosp,  // [B,S,64]
    const float* __restrict__ sinp,  // [B,S,64]
    float* __restrict__ qf, float* __restrict__ kf, float* __restrict__ vf)
{
    const int s = blockIdx.x, b = blockIdx.y;
    const long long bs = (long long)b * SS + s;
    const float* qrow  = qg  + bs * (NH * D_QK);
    const float* kvrow = kvg + bs * (NH * (D_NOPE + D_V));
    const float* crow  = ckv + bs * (KV_LORA + D_ROPE);
    const float* c  = cosp + bs * D_ROPE;
    const float* sn = sinp + bs * D_ROPE;

    for (int i = threadIdx.x; i < NH * D_QK; i += 256) {
        const int h = i / D_QK, d = i % D_QK;
        float v;
        if (d < D_NOPE) {
            v = qrow[h * D_QK + d];
        } else {
            const int dr = d - D_NOPE;
            const float x  = qrow[h * D_QK + D_NOPE + dr];
            const float xr = (dr < 32) ? -qrow[h * D_QK + D_NOPE + dr + 32]
                                       :  qrow[h * D_QK + D_NOPE + dr - 32];
            v = x * c[dr] + xr * sn[dr];
        }
        qf[(((long long)b * NH + h) * SS + s) * D_QK + d] = v;
    }
    for (int i = threadIdx.x; i < NH * D_QK; i += 256) {
        const int h = i / D_QK, d = i % D_QK;
        float v;
        if (d < D_NOPE) {
            v = kvrow[h * (D_NOPE + D_V) + d];
        } else {
            const int dr = d - D_NOPE;
            const float x  = crow[KV_LORA + dr];
            const float xr = (dr < 32) ? -crow[KV_LORA + dr + 32]
                                       :  crow[KV_LORA + dr - 32];
            v = x * c[dr] + xr * sn[dr];
        }
        kf[(((long long)b * NH + h) * SS + s) * D_QK + d] = v;
    }
    for (int i = threadIdx.x; i < NH * D_V; i += 256) {
        const int h = i / D_V, d = i % D_V;
        vf[(((long long)b * NH + h) * SS + s) * D_V + d] =
            kvrow[h * (D_NOPE + D_V) + D_NOPE + d];
    }
}

// ---------------------------------------------------------------------------
// In-place masked softmax over rows of attn [B*H,S,S]. grid = (S, B*H).
// ---------------------------------------------------------------------------
__global__ __launch_bounds__(256) void softmax_mask_kernel(
    float* __restrict__ attn, const float* __restrict__ mask)
{
    const int q = blockIdx.x, bh = blockIdx.y, b = bh / NH;
    float* row        = attn + (((long long)bh) * SS + q) * SS;
    const float* mrow = mask + (((long long)b)  * SS + q) * SS;
    __shared__ float red[256];

    float vals[SS / 256];
    float mx = -3.4e38f;
    #pragma unroll
    for (int j = 0; j < SS / 256; ++j) {
        const int k = threadIdx.x + j * 256;
        const float v = row[k] + mrow[k];
        vals[j] = v; mx = fmaxf(mx, v);
    }
    red[threadIdx.x] = mx; __syncthreads();
    for (int off = 128; off > 0; off >>= 1) {
        if ((int)threadIdx.x < off)
            red[threadIdx.x] = fmaxf(red[threadIdx.x], red[threadIdx.x + off]);
        __syncthreads();
    }
    mx = red[0]; __syncthreads();

    float sum = 0.f;
    #pragma unroll
    for (int j = 0; j < SS / 256; ++j) { float e = __expf(vals[j] - mx); vals[j] = e; sum += e; }
    red[threadIdx.x] = sum; __syncthreads();
    for (int off = 128; off > 0; off >>= 1) {
        if ((int)threadIdx.x < off) red[threadIdx.x] += red[threadIdx.x + off];
        __syncthreads();
    }
    const float inv = 1.f / red[0];
    #pragma unroll
    for (int j = 0; j < SS / 256; ++j) row[threadIdx.x + j * 256] = vals[j] * inv;
}

// ---------------------------------------------------------------------------
// Merge heads: out_heads [B*H,S,128] -> merged [B*S, H*128]. grid = (S, B)
// ---------------------------------------------------------------------------
__global__ __launch_bounds__(256) void merge_heads_kernel(
    const float* __restrict__ oh, float* __restrict__ merged)
{
    const int s = blockIdx.x, b = blockIdx.y;
    for (int i = threadIdx.x; i < NH * D_V; i += 256) {
        const int h = i >> 7, d = i & 127;
        merged[((long long)b * SS + s) * (NH * D_V) + h * D_V + d] =
            oh[(((long long)b * NH + h) * SS + s) * D_V + d];
    }
}

// ---------------------------------------------------------------------------
static void launch_gemm(hipStream_t stream,
                        const float* A, const float* Bm, float* C,
                        int M, int N, int K, int lda, int ldb, int ldc,
                        long long sA, long long sB, long long sC,
                        int batch, int b_is_NK, float alpha)
{
    dim3 grid(N / 64, M / 64, batch);
    gemm_bf16_wmma<<<grid, 256, 0, stream>>>(A, Bm, C, M, N, K, lda, ldb, ldc,
                                             sA, sB, sC, b_is_NK, alpha);
}

extern "C" void kernel_launch(void* const* d_in, const int* in_sizes, int n_in,
                              void* d_out, int out_size, void* d_ws, size_t ws_size,
                              hipStream_t stream)
{
    (void)in_sizes; (void)n_in; (void)out_size; (void)ws_size;

    const float* hidden   = (const float*)d_in[0];   // [B,S,HID]
    const float* cosp     = (const float*)d_in[1];   // [B,S,64]
    const float* sinp     = (const float*)d_in[2];   // [B,S,64]
    const float* mask     = (const float*)d_in[3];   // [B,1,S,S]
    const float* q_a_w    = (const float*)d_in[4];   // [HID,Q_LORA]
    const float* q_a_ln_w = (const float*)d_in[5];   // [Q_LORA]
    const float* q_b_w    = (const float*)d_in[6];   // [Q_LORA,H*D_QK]
    const float* kv_a_w   = (const float*)d_in[7];   // [HID,KV_LORA+D_ROPE]
    const float* kv_a_ln  = (const float*)d_in[8];   // [KV_LORA]
    const float* kv_b_w   = (const float*)d_in[9];   // [KV_LORA,H*(D_NOPE+D_V)]
    const float* o_w      = (const float*)d_in[10];  // [H*D_V,HID]

    float* attn_out = (float*)d_out;                         // [B,S,H*D_V]
    float* attn_w   = attn_out + (long long)BB * SS * NH * D_V;  // [B,H,S,S]

    const long long M_rows = (long long)BB * SS;   // 4096

    // workspace layout (floats)
    float* ws = (float*)d_ws;
    float* q_lin     = ws;                                      // 4096x1536
    float* q_gemm    = q_lin     + M_rows * Q_LORA;             // 4096x3072
    float* ckv       = q_gemm    + M_rows * (NH * D_QK);        // 4096x576
    float* kv        = ckv       + M_rows * (KV_LORA + D_ROPE); // 4096x4096
    float* q_full    = kv        + M_rows * (NH * (D_NOPE + D_V)); // 32x2048x192
    float* k_full    = q_full    + (long long)BB * NH * SS * D_QK;
    float* v_heads   = k_full    + (long long)BB * NH * SS * D_QK; // 32x2048x128
    float* out_heads = v_heads   + (long long)BB * NH * SS * D_V;  // 32x2048x128
    float* merged    = q_full;   // reuse: q_full dead after QK^T

    const float scaling = 1.0f / sqrtf((float)D_QK);

    // 1) q_lin = hidden @ q_a_w          [4096,1536]
    launch_gemm(stream, hidden, q_a_w, q_lin,
                (int)M_rows, Q_LORA, HID, HID, Q_LORA, Q_LORA, 0, 0, 0, 1, 0, 1.0f);
    // 2) q_resid = rms_norm(q_lin)       (in place)
    rmsnorm_kernel<<<(int)M_rows, 256, 0, stream>>>(q_lin, q_a_ln_w, q_lin,
                                                    Q_LORA, Q_LORA, Q_LORA);
    // 3) q = q_resid @ q_b_w             [4096,3072]
    launch_gemm(stream, q_lin, q_b_w, q_gemm,
                (int)M_rows, NH * D_QK, Q_LORA, Q_LORA, NH * D_QK, NH * D_QK,
                0, 0, 0, 1, 0, 1.0f);
    // 4) ckv = hidden @ kv_a_w           [4096,576]
    launch_gemm(stream, hidden, kv_a_w, ckv,
                (int)M_rows, KV_LORA + D_ROPE, HID, HID,
                KV_LORA + D_ROPE, KV_LORA + D_ROPE, 0, 0, 0, 1, 0, 1.0f);
    // 5) k_c = rms_norm(ckv[..., :512])  (in place on 512-wide slice)
    rmsnorm_kernel<<<(int)M_rows, 256, 0, stream>>>(ckv, kv_a_ln, ckv,
                                                    KV_LORA, KV_LORA + D_ROPE,
                                                    KV_LORA + D_ROPE);
    // 6) kv = k_c @ kv_b_w               [4096,4096] (A lda=576, K=512)
    launch_gemm(stream, ckv, kv_b_w, kv,
                (int)M_rows, NH * (D_NOPE + D_V), KV_LORA, KV_LORA + D_ROPE,
                NH * (D_NOPE + D_V), NH * (D_NOPE + D_V), 0, 0, 0, 1, 0, 1.0f);
    // 7) pack heads + RoPE
    {
        dim3 grid(SS, BB);
        build_qkv_kernel<<<grid, 256, 0, stream>>>(q_gemm, kv, ckv, cosp, sinp,
                                                   q_full, k_full, v_heads);
    }
    // 8) logits = scaling * Q @ K^T  -> attn_w  (batched over B*H, B is [N,K])
    launch_gemm(stream, q_full, k_full, attn_w,
                SS, SS, D_QK, D_QK, D_QK, SS,
                (long long)SS * D_QK, (long long)SS * D_QK, (long long)SS * SS,
                BB * NH, 1, scaling);
    // 9) softmax(logits + mask) in place
    {
        dim3 grid(SS, BB * NH);
        softmax_mask_kernel<<<grid, 256, 0, stream>>>(attn_w, mask);
    }
    // 10) out_heads = P @ V              (batched over B*H)
    launch_gemm(stream, attn_w, v_heads, out_heads,
                SS, D_V, SS, SS, D_V, D_V,
                (long long)SS * SS, (long long)SS * D_V, (long long)SS * D_V,
                BB * NH, 0, 1.0f);
    // 11) merge heads -> [4096, 2048]
    {
        dim3 grid(SS, BB);
        merge_heads_kernel<<<grid, 256, 0, stream>>>(out_heads, merged);
    }
    // 12) attn_out = merged @ o_w        [4096,2048]
    launch_gemm(stream, merged, o_w, attn_out,
                (int)M_rows, HID, NH * D_V, NH * D_V, HID, HID, 0, 0, 0, 1, 0, 1.0f);
}
